// NSAAttentionExtendedWithRouting_52527450030124
// MI455X (gfx1250) — compile-verified, matrix operand order checked by reading, not run on hardware
//
#include <hip/hip_runtime.h>

#define T_TOK 8192
#define DIM   1024
#define DFF_  4096
#define NEXP  4
#define NSH   2
#define CAP   2560

typedef __attribute__((ext_vector_type(16))) __bf16 v16bf;
typedef __attribute__((ext_vector_type(8)))  float  v8f;

union FragAB { v16bf v; uint4 q[2]; };

__device__ __forceinline__ unsigned short f2bf(float f) {
  unsigned int u = __float_as_uint(f);
  u += 0x7FFFu + ((u >> 16) & 1u);           // round-to-nearest-even
  return (unsigned short)(u >> 16);
}
__device__ __forceinline__ float gelu_exact(float v) {
  return 0.5f * v * (1.0f + erff(v * 0.70710678118654752440f));
}

// ---------------------------------------------------------------------------
// 256x128-tile WMMA GEMM, double-buffered LDS, B pre-transposed to [N][K].
// C = epi( gather(A) @ B[seg]^T + bias[seg] )
// epi: 0 = gelu -> bf16 store, 1 = gelu -> f32 store, 2 = f32 store,
//      3 = scatter-add  final[tok] += w * (acc + bias), skipping pad tokens.
// Epilogue is staged through LDS (64x128 f32 chunks) so accumulators never
// spill to scratch and all global stores are wide and coalesced.
// ---------------------------------------------------------------------------
__global__ __launch_bounds__(256) void wmma_gemm(
    const unsigned short* __restrict__ A, int lda,
    const unsigned short* __restrict__ BT, int ldb, long segB,
    const float* __restrict__ bias, int segBias,
    int K,
    void* __restrict__ Cv, int ldc,
    const int* __restrict__ rowmap,
    const float* __restrict__ roww, float wconst,
    int epi, int gatherA, int segM)
{
  __shared__ unsigned short lA[2][256 * 40];   // [row][k], padded (80B rows, 16B aligned)
  __shared__ unsigned short lB[2][128 * 40];   // [n][k], padded

  const int tid = threadIdx.x, lane = tid & 31, wid = tid >> 5;
  const int wm = wid & 3, wn = wid >> 2;       // wave tile: 64 rows x 64 cols
  const int half = lane >> 4, l15 = lane & 15;

  const long row0 = (long)blockIdx.z * segM + (long)blockIdx.x * 256;
  const long col0 = (long)blockIdx.y * 128;
  const unsigned short* Bseg = BT + (long)blockIdx.z * segB;
  const float* biasseg = bias + (long)blockIdx.z * segBias;

  // staging assignments (256 threads)
  const int arow = tid;                        // A: 256 rows x 32 k (one row each)
  const int brow = tid >> 1, bks = (tid & 1) * 16;  // B: 128 n-rows x 32 k (half row each)

  long atok = row0 + arow;
  if (gatherA) atok = (long)rowmap[atok];
  const bool avalid = (!gatherA) || (atok < (long)T_TOK);
  const unsigned short* Arow = A + atok * (long)lda;
  const unsigned short* Brow = Bseg + (long)(col0 + brow) * ldb + bks;

  const int nIter = K >> 5;
  uint4 ar0, ar1, ar2, ar3, br0, br1;

  // prologue: tile 0 -> regs -> LDS buf 0
  ar0 = ar1 = ar2 = ar3 = make_uint4(0, 0, 0, 0);
  if (avalid) {
    const uint4* p = (const uint4*)Arow;
    ar0 = p[0]; ar1 = p[1]; ar2 = p[2]; ar3 = p[3];
  }
  {
    const uint4* q = (const uint4*)Brow;
    br0 = q[0]; br1 = q[1];
  }
  {
    uint4* da = (uint4*)&lA[0][arow * 40];
    da[0] = ar0; da[1] = ar1; da[2] = ar2; da[3] = ar3;
    uint4* db = (uint4*)&lB[0][brow * 40 + bks];
    db[0] = br0; db[1] = br1;
  }
  __syncthreads();

  v8f acc[4][4] = {};

  for (int it = 0; it < nIter; ++it) {
    const int cur = it & 1;
    if (it + 1 < nIter) {               // global -> regs for next tile
      const int kk = (it + 1) << 5;
      ar0 = ar1 = ar2 = ar3 = make_uint4(0, 0, 0, 0);
      if (avalid) {
        const uint4* p = (const uint4*)(Arow + kk);
        ar0 = p[0]; ar1 = p[1]; ar2 = p[2]; ar3 = p[3];
      }
      const uint4* q = (const uint4*)(Brow + kk);
      br0 = q[0]; br1 = q[1];
      if (it + 2 < nIter) {             // -> global_prefetch_b8
        if (avalid) __builtin_prefetch(Arow + kk + 32, 0, 1);
        __builtin_prefetch(Brow + kk + 32, 0, 1);
      }
    }

    FragAB bfr[4];
#pragma unroll
    for (int ni = 0; ni < 4; ++ni) {
      const int n = wn * 64 + ni * 16 + l15;
      bfr[ni].q[0] = *(const uint4*)&lB[cur][n * 40 + 8 * half];
      bfr[ni].q[1] = *(const uint4*)&lB[cur][n * 40 + 16 + 8 * half];
    }
#pragma unroll
    for (int mi = 0; mi < 4; ++mi) {
      const int m = wm * 64 + mi * 16 + l15;
      FragAB af;
      af.q[0] = *(const uint4*)&lA[cur][m * 40 + 8 * half];
      af.q[1] = *(const uint4*)&lA[cur][m * 40 + 16 + 8 * half];
#pragma unroll
      for (int ni = 0; ni < 4; ++ni)
        acc[mi][ni] = __builtin_amdgcn_wmma_f32_16x16x32_bf16(
            false, af.v, false, bfr[ni].v, (short)0, acc[mi][ni], false, false);
    }

    if (it + 1 < nIter) {               // regs -> LDS other buffer
      uint4* da = (uint4*)&lA[cur ^ 1][arow * 40];
      da[0] = ar0; da[1] = ar1; da[2] = ar2; da[3] = ar3;
      uint4* db = (uint4*)&lB[cur ^ 1][brow * 40 + bks];
      db[0] = br0; db[1] = br1;
    }
    __syncthreads();
  }

  // ---- epilogue: stage 64x128 f32 chunks through LDS, coalesced stores ----
  float* lC = (float*)&lA[0][0];        // 64 x 132 f32 = 33792B <= 40960B
  const int lrow = tid >> 2;            // reader: one row, 32 contiguous cols
  const int seg32 = (tid & 3) * 32;

#pragma unroll
  for (int mi = 0; mi < 4; mi++) {
    // writers: this wave's 16x64 sub-tile for chunk mi
#pragma unroll
    for (int ni = 0; ni < 4; ni++) {
#pragma unroll
      for (int i = 0; i < 8; i++)
        lC[(wm * 16 + i + 8 * half) * 132 + wn * 64 + ni * 16 + l15] = acc[mi][ni][i];
    }
    __syncthreads();
    {
      const long gr = row0 + (long)(lrow >> 4) * 64 + mi * 16 + (lrow & 15);
      const float* lp = lC + lrow * 132 + seg32;
      const float* bp = biasseg + col0 + seg32;
      if (epi == 0) {
        unsigned int ov[16];
#pragma unroll
        for (int e = 0; e < 16; e++) {
          float v0 = gelu_exact(lp[2 * e] + bp[2 * e]);
          float v1 = gelu_exact(lp[2 * e + 1] + bp[2 * e + 1]);
          ov[e] = (unsigned int)f2bf(v0) | ((unsigned int)f2bf(v1) << 16);
        }
        uint4* dst = (uint4*)((unsigned short*)Cv + gr * ldc + col0 + seg32);
        dst[0] = *(uint4*)&ov[0];  dst[1] = *(uint4*)&ov[4];
        dst[2] = *(uint4*)&ov[8];  dst[3] = *(uint4*)&ov[12];
      } else if (epi == 1) {
        float* dst = (float*)Cv + gr * ldc + col0 + seg32;
#pragma unroll
        for (int e = 0; e < 32; e++) dst[e] = gelu_exact(lp[e] + bp[e]);
      } else if (epi == 2) {
        float* dst = (float*)Cv + gr * ldc + col0 + seg32;
#pragma unroll
        for (int e = 0; e < 32; e++) dst[e] = lp[e] + bp[e];
      } else {
        long tok = rowmap ? (long)rowmap[gr] : gr;
        if (tok < (long)T_TOK) {
          float wgt = roww ? roww[gr] : wconst;
          float* dst = (float*)Cv + tok * ldc + col0 + seg32;
#pragma unroll
          for (int e = 0; e < 32; e++) dst[e] += wgt * (lp[e] + bp[e]);
        }
      }
    }
    __syncthreads();
  }
}

// ---------------------------------------------------------------------------
__global__ void cvt_bf16(const float* __restrict__ s, unsigned short* __restrict__ d, long n) {
  long i = (long)blockIdx.x * blockDim.x + threadIdx.x;
  long st = (long)gridDim.x * blockDim.x;
  for (; i < n; i += st) d[i] = f2bf(s[i]);
}

// fp32 [K][N] -> bf16 [N][K] (per-segment), tiled 32x32 transpose through LDS
__global__ void cvt_tr(const float* __restrict__ src, unsigned short* __restrict__ dst,
                       int K, int N) {
  __shared__ float t[32][33];
  const long base = (long)blockIdx.z * K * N;
  const int n0 = blockIdx.x * 32, k0 = blockIdx.y * 32;
  const int tx = threadIdx.x & 31, ty = threadIdx.x >> 5;   // 32 x 8
  const float* s = src + base;
  unsigned short* d = dst + base;
#pragma unroll
  for (int r = ty; r < 32; r += 8)
    t[r][tx] = s[(long)(k0 + r) * N + n0 + tx];
  __syncthreads();
#pragma unroll
  for (int r = ty; r < 32; r += 8)
    d[(long)(n0 + r) * K + k0 + tx] = f2bf(t[tx][r]);
}

__global__ void router2_k(const float* __restrict__ hrt, const float* __restrict__ w2,
                          const float* __restrict__ b2, float* __restrict__ logits) {
  int lane = threadIdx.x & 31, wid = threadIdx.x >> 5;
  long t = (long)blockIdx.x * 8 + wid;
  float a0 = 0, a1 = 0, a2 = 0, a3 = 0;
  for (int k = lane; k < DIM; k += 32) {
    float h = hrt[t * DIM + k];
    a0 += h * w2[k * NEXP + 0]; a1 += h * w2[k * NEXP + 1];
    a2 += h * w2[k * NEXP + 2]; a3 += h * w2[k * NEXP + 3];
  }
#pragma unroll
  for (int off = 16; off; off >>= 1) {
    a0 += __shfl_xor(a0, off, 32); a1 += __shfl_xor(a1, off, 32);
    a2 += __shfl_xor(a2, off, 32); a3 += __shfl_xor(a3, off, 32);
  }
  if (lane == 0) {
    logits[t * NEXP + 0] = a0 + b2[0]; logits[t * NEXP + 1] = a1 + b2[1];
    logits[t * NEXP + 2] = a2 + b2[2]; logits[t * NEXP + 3] = a3 + b2[3];
  }
}

__global__ void routing_k(const float* __restrict__ logits, int* __restrict__ topi,
                          float* __restrict__ topw, float* __restrict__ part) {
  long t = (long)blockIdx.x * 256 + threadIdx.x;
  float l[NEXP];
#pragma unroll
  for (int e = 0; e < NEXP; e++) l[e] = logits[t * NEXP + e];
  int i0 = 0; float v0 = l[0];
#pragma unroll
  for (int e = 1; e < NEXP; e++) if (l[e] > v0) { v0 = l[e]; i0 = e; }
  int i1 = -1; float v1 = -3.4e38f;
#pragma unroll
  for (int e = 0; e < NEXP; e++) if (e != i0 && l[e] > v1) { v1 = l[e]; i1 = e; }
  topi[t * 2 + 0] = i0; topi[t * 2 + 1] = i1;
  float e1 = expf(v1 - v0);
  float s = 1.0f + e1;
  topw[t * 2 + 0] = 1.0f / s;
  topw[t * 2 + 1] = e1 / s;
  float m4 = fmaxf(fmaxf(l[0], l[1]), fmaxf(l[2], l[3]));
  float se = 0.f;
#pragma unroll
  for (int e = 0; e < NEXP; e++) se += expf(l[e] - m4);
  float lse = m4 + logf(se);
  __shared__ float red[256];
  red[threadIdx.x] = lse * lse;
  __syncthreads();
  for (int o = 128; o; o >>= 1) {
    if (threadIdx.x < o) red[threadIdx.x] += red[threadIdx.x + o];
    __syncthreads();
  }
  if (threadIdx.x == 0) part[blockIdx.x] = red[0];
}

__global__ void zfinal_k(const float* __restrict__ part, int nb, float* __restrict__ zout) {
  float s = 0.f;
  for (int i = 0; i < nb; i++) s += part[i];
  *zout = s / (float)T_TOK;
}

// One wave per (slot,expert): ordered, capacity-capped compaction.
__global__ void build_lists_k(const int* __restrict__ topi, const float* __restrict__ topw,
                              int* __restrict__ rtok, float* __restrict__ rwt) {
  int seg = blockIdx.x;                 // slot*NEXP + j
  int slot = seg >> 2, j = seg & 3;
  int lane = threadIdx.x;
  int base = 0;
  for (int t0 = 0; t0 < T_TOK && base < CAP; t0 += 32) {
    int tok = t0 + lane;
    bool pred = (topi[tok * 2 + slot] == j);
    unsigned int m = __builtin_amdgcn_ballot_w32(pred);
    int pos = base + __popc(m & ((1u << lane) - 1u));
    if (pred && pos < CAP) {
      rtok[seg * CAP + pos] = tok;
      rwt[seg * CAP + pos] = topw[tok * 2 + slot];
    }
    base += __popc(m);
  }
  if (base > CAP) base = CAP;
  for (int p = base + lane; p < CAP; p += 32) {
    rtok[seg * CAP + p] = T_TOK;        // pad token -> zero row / skip scatter
    rwt[seg * CAP + p] = 0.f;
  }
}

__global__ void ln_k(const float* __restrict__ outpre, const float* __restrict__ x,
                     float* __restrict__ out) {
  long t = blockIdx.x;
  int tid = threadIdx.x;
  float v[4]; float s = 0.f, s2 = 0.f;
#pragma unroll
  for (int i = 0; i < 4; i++) {
    int dcol = tid + i * 256;
    float ov = 0.5f * outpre[t * DIM + dcol] + 0.5f * x[t * DIM + dcol];
    v[i] = ov; s += ov; s2 += ov * ov;
  }
  __shared__ float r1[256], r2[256];
  r1[tid] = s; r2[tid] = s2;
  __syncthreads();
  for (int o = 128; o; o >>= 1) {
    if (tid < o) { r1[tid] += r1[tid + o]; r2[tid] += r2[tid + o]; }
    __syncthreads();
  }
  float mu = r1[0] * (1.f / DIM);
  float var = r2[0] * (1.f / DIM) - mu * mu;
  float inv = rsqrtf(var + 1e-6f);
#pragma unroll
  for (int i = 0; i < 4; i++) {
    int dcol = tid + i * 256;
    out[t * DIM + dcol] = (v[i] - mu) * inv;
  }
}

// ---------------------------------------------------------------------------
extern "C" void kernel_launch(void* const* d_in, const int* in_sizes, int n_in,
                              void* d_out, int out_size, void* d_ws, size_t ws_size,
                              hipStream_t stream) {
  (void)in_sizes; (void)n_in; (void)out_size; (void)ws_size;
  const float* x     = (const float*)d_in[0];
  const float* rw1   = (const float*)d_in[1];
  const float* rb1   = (const float*)d_in[2];
  const float* rw2   = (const float*)d_in[3];
  const float* rb2   = (const float*)d_in[4];
  const float* re_w1 = (const float*)d_in[5];
  const float* re_b1 = (const float*)d_in[6];
  const float* re_w2 = (const float*)d_in[7];
  const float* re_b2 = (const float*)d_in[8];
  const float* se_w1 = (const float*)d_in[9];
  const float* se_b1 = (const float*)d_in[10];
  const float* se_w2 = (const float*)d_in[11];
  const float* se_b2 = (const float*)d_in[12];
  const float* out_w = (const float*)d_in[13];
  const float* out_b = (const float*)d_in[14];
  float* out = (float*)d_out;

  char* w = (char*)d_ws;
  size_t off = 0;
  auto alloc = [&](size_t bytes) -> char* {
    char* p = w + off; off = (off + bytes + 255) & ~(size_t)255; return p;
  };
  unsigned short* xb    = (unsigned short*)alloc((size_t)T_TOK * DIM * 2);
  unsigned short* wr1b  = (unsigned short*)alloc((size_t)DIM * DIM * 2);          // [N][K]
  unsigned short* wre1b = (unsigned short*)alloc((size_t)NEXP * DIM * DFF_ * 2);  // [N=DFF][K=DIM]
  unsigned short* wre2b = (unsigned short*)alloc((size_t)NEXP * DFF_ * DIM * 2);  // [N=DIM][K=DFF]
  unsigned short* wse1b = (unsigned short*)alloc((size_t)NSH * DIM * DFF_ * 2);
  unsigned short* wse2b = (unsigned short*)alloc((size_t)NSH * DFF_ * DIM * 2);
  unsigned short* wob   = (unsigned short*)alloc((size_t)DIM * DIM * 2);
  float* hrt    = (float*)alloc((size_t)T_TOK * DIM * 4);   // router hidden, reused as outpre
  float* logits = (float*)alloc((size_t)T_TOK * NEXP * 4);
  int*   topi   = (int*)alloc((size_t)T_TOK * 2 * 4);
  float* topw   = (float*)alloc((size_t)T_TOK * 2 * 4);
  float* part   = (float*)alloc(64 * 4);
  int*   rtok   = (int*)alloc((size_t)8 * CAP * 4);
  float* rwt    = (float*)alloc((size_t)8 * CAP * 4);
  unsigned short* h1 = (unsigned short*)alloc((size_t)(NEXP * CAP) * DFF_ * 2); // 10240 x 4096 bf16
  float* finalb = (float*)alloc((size_t)T_TOK * DIM * 4);
  unsigned short* fb = (unsigned short*)alloc((size_t)T_TOK * DIM * 2);

  // activations: plain fp32->bf16; weights: fp32 [K][N] -> bf16 [N][K]
  cvt_bf16<<<2048, 256, 0, stream>>>(x, xb, (long)T_TOK * DIM);
  cvt_tr<<<dim3(DIM / 32, DIM / 32, 1), 256, 0, stream>>>(rw1, wr1b, DIM, DIM);
  cvt_tr<<<dim3(DFF_ / 32, DIM / 32, NEXP), 256, 0, stream>>>(re_w1, wre1b, DIM, DFF_);
  cvt_tr<<<dim3(DIM / 32, DFF_ / 32, NEXP), 256, 0, stream>>>(re_w2, wre2b, DFF_, DIM);
  cvt_tr<<<dim3(DFF_ / 32, DIM / 32, NSH), 256, 0, stream>>>(se_w1, wse1b, DIM, DFF_);
  cvt_tr<<<dim3(DIM / 32, DFF_ / 32, NSH), 256, 0, stream>>>(se_w2, wse2b, DFF_, DIM);
  cvt_tr<<<dim3(DIM / 32, DIM / 32, 1), 256, 0, stream>>>(out_w, wob, DIM, DIM);

  hipMemsetAsync(finalb, 0, (size_t)T_TOK * DIM * 4, stream);

  // router: hrt = gelu(x@W1+b1);  logits = hrt@W2+b2
  wmma_gemm<<<dim3(T_TOK / 256, DIM / 128, 1), 256, 0, stream>>>(
      xb, DIM, wr1b, DIM, 0, rb1, 0, DIM, hrt, DIM,
      nullptr, nullptr, 0.f, /*epi=*/1, /*gatherA=*/0, T_TOK);
  router2_k<<<T_TOK / 8, 256, 0, stream>>>(hrt, rw2, rb2, logits);
  routing_k<<<T_TOK / 256, 256, 0, stream>>>(logits, topi, topw, part);
  zfinal_k<<<1, 1, 0, stream>>>(part, T_TOK / 256, out + (size_t)T_TOK * DIM);
  build_lists_k<<<8, 32, 0, stream>>>(topi, topw, rtok, rwt);

  // routed experts, one slot at a time (tokens unique within slot -> race-free scatter)
  for (int slot = 0; slot < 2; slot++) {
    const int* rm = rtok + (size_t)slot * NEXP * CAP;
    const float* rw = rwt + (size_t)slot * NEXP * CAP;
    wmma_gemm<<<dim3(CAP / 256, DFF_ / 128, NEXP), 256, 0, stream>>>(
        xb, DIM, wre1b, DIM, (long)DIM * DFF_, re_b1, DFF_, DIM,
        h1, DFF_, rm, nullptr, 0.f, /*epi=*/0, /*gatherA=*/1, CAP);
    wmma_gemm<<<dim3(CAP / 256, DIM / 128, NEXP), 256, 0, stream>>>(
        h1, DFF_, wre2b, DFF_, (long)DFF_ * DIM, re_b2, DIM, DFF_,
        finalb, DIM, rm, rw, 0.f, /*epi=*/3, /*gatherA=*/0, CAP);
  }
  // shared experts over all tokens, weight 1/NS
  for (int j = 0; j < NSH; j++) {
    wmma_gemm<<<dim3(T_TOK / 256, DFF_ / 128, 1), 256, 0, stream>>>(
        xb, DIM, wse1b + (size_t)j * DIM * DFF_, DIM, 0,
        se_b1 + (size_t)j * DFF_, 0, DIM,
        h1, DFF_, nullptr, nullptr, 0.f, /*epi=*/0, /*gatherA=*/0, T_TOK);
    wmma_gemm<<<dim3(T_TOK / 256, DIM / 128, 1), 256, 0, stream>>>(
        h1, DFF_, wse2b + (size_t)j * DFF_ * DIM, DFF_, 0,
        se_b2 + (size_t)j * DIM, 0, DFF_,
        finalb, DIM, nullptr, nullptr, 0.5f, /*epi=*/3, /*gatherA=*/0, T_TOK);
  }
  // output projection + residual mix + LN
  cvt_bf16<<<2048, 256, 0, stream>>>(finalb, fb, (long)T_TOK * DIM);
  wmma_gemm<<<dim3(T_TOK / 256, DIM / 128, 1), 256, 0, stream>>>(
      fb, DIM, wob, DIM, 0, out_b, 0, DIM, hrt, DIM,
      nullptr, nullptr, 0.f, /*epi=*/2, /*gatherA=*/0, T_TOK);
  ln_k<<<T_TOK, 256, 0, stream>>>(hrt, x, out);
}
